// FileEncoder_34007551049873
// MI455X (gfx1250) — matrix operand order repaired and checked by reading, not compiled, not run on hardware
//
#include <hip/hip_runtime.h>

typedef __attribute__((ext_vector_type(16))) _Float16 v16h;
typedef __attribute__((ext_vector_type(8)))  _Float16 v8h;
typedef __attribute__((ext_vector_type(8)))  float    v8f;

#define DMODEL 256
#define SEQ    64
#define NHEAD  8
#define HDIM   32
#define NLAYER 4
#define FFN    1024
#define NEGINF (-1e9f)

__device__ __forceinline__ v8f vzero8() { v8f z = {}; return z; }

// A fragment (16-bit A, 16x32 MxK): lane<16 -> m=lane, kb=0 ; lane>=16 -> m=lane-16, kb=16
// elements 0..15 = K kb..kb+15
__device__ __forceinline__ v16h load_a_frag(const _Float16* A, int lda, int mB, int kB, int lane) {
  int m  = mB + (lane & 15);
  int kb = (lane >> 4) * 16;
  const _Float16* p = A + m * lda + kB + kb;
  v8h lo = *(const v8h*)(p);
  v8h hi = *(const v8h*)(p + 8);
  v16h r;
#pragma unroll
  for (int j = 0; j < 8; ++j) { r[j] = lo[j]; r[8 + j] = hi[j]; }
  return r;
}

// B fragment (16-bit B, 32x16 KxN) read from "transposed" storage Bt[n][k] (row n contiguous in k):
// lane -> n = lane&15 ; elements j -> k = (lane>>4)*16 + j
__device__ __forceinline__ v16h load_b_frag(const _Float16* Bt, int ldk, int nB, int kB, int lane) {
  const _Float16* p = Bt + (nB + (lane & 15)) * ldk + kB + ((lane >> 4) << 4);
  return *(const v16h*)p;
}

template <int KLEN>
__device__ __forceinline__ v8f mma_acc_t(const _Float16* A, int lda, const _Float16* Bt, int ldk,
                                         int mB, int nB, int kOff, int lane, v8f acc) {
  __builtin_prefetch(Bt + (nB + (lane & 15)) * ldk + kOff, 0, 0);
#pragma unroll
  for (int k = 0; k < KLEN; k += 32) {
    v16h a = load_a_frag(A, lda, mB, k, lane);
    v16h b = load_b_frag(Bt, ldk, nB, kOff + k, lane);
    acc = __builtin_amdgcn_wmma_f32_16x16x32_f16(false, a, false, b, (short)0, acc, false, false);
  }
  return acc;
}

__device__ __forceinline__ v8f mma256(const _Float16* A, int lda, const _Float16* Bt, int ldk,
                                      int mB, int nB, int lane) {
  return mma_acc_t<256>(A, lda, Bt, ldk, mB, nB, 0, lane, vzero8());
}

// post-norm LN in place over f16 rows; wave w owns rows w*8..w*8+7, lane owns 8 cols
__device__ __forceinline__ void layernorm_inplace(_Float16* sR, const float* gs, const float* gb,
                                                  int w, int lane) {
#pragma unroll
  for (int rr = 0; rr < 8; ++rr) {
    int row = w * 8 + rr;
    _Float16* p = sR + row * DMODEL + lane * 8;
    float v[8];
    float s = 0.f;
#pragma unroll
    for (int j = 0; j < 8; ++j) { v[j] = (float)p[j]; s += v[j]; }
#pragma unroll
    for (int off = 1; off < 32; off <<= 1) s += __shfl_xor(s, off, 32);
    float mean = s * (1.0f / 256.0f);
    float var = 0.f;
#pragma unroll
    for (int j = 0; j < 8; ++j) { float d = v[j] - mean; var += d * d; }
#pragma unroll
    for (int off = 1; off < 32; off <<= 1) var += __shfl_xor(var, off, 32);
    float rs = rsqrtf(var * (1.0f / 256.0f) + 1e-5f);
#pragma unroll
    for (int j = 0; j < 8; ++j) {
      int col = lane * 8 + j;
      p[j] = (_Float16)((v[j] - mean) * rs * gs[col] + gb[col]);
    }
  }
}

// weights f32 [L][K][N] -> f16 transposed [L][N][K]
__global__ void wcast_transpose(const float* __restrict__ src, _Float16* __restrict__ dst,
                                int L, int K, int N) {
  long long idx = (long long)blockIdx.x * blockDim.x + threadIdx.x;
  long long total = (long long)L * K * N;
  if (idx >= total) return;
  long long per = (long long)K * N;
  long long l = idx / per;
  long long r = idx - l * per;
  long long n = r / K;
  long long k = r - n * K;
  dst[idx] = (_Float16)src[l * per + k * N + n];
}

// scope_reprs init = sinusoidal pos enc, base freq 500
__global__ void pos_enc_kernel(const int* __restrict__ pos, float* __restrict__ out) {
  int n = blockIdx.x, d = threadIdx.x;
  float p = (float)pos[n];
  int i = d & 127;
  float ang = p * __expf(-(float)i * (1.0f / 128.0f) * __logf(500.0f));
  out[n * DMODEL + d] = (d < 128) ? __sinf(ang) : __cosf(ang);
}

__global__ void rank_update_kernel(const int* __restrict__ sortk, const float* __restrict__ cls,
                                   float* __restrict__ out, int rank) {
  int n = blockIdx.x, d = threadIdx.x;
  if (sortk[n] == rank) out[n * DMODEL + d] += cls[n * DMODEL + d];
}

// one block = one sequence; 8 waves; everything in LDS; f16 residual, f32 WMMA accumulate
__global__ __launch_bounds__(256) void encoder_kernel(
    const int* __restrict__ tokens, const unsigned char* __restrict__ pad,
    const unsigned char* __restrict__ refmask, const float* __restrict__ emb,
    const float* __restrict__ storage,
    const _Float16* __restrict__ WqT, const _Float16* __restrict__ WkT,
    const _Float16* __restrict__ WvT, const _Float16* __restrict__ WoT,
    const float* __restrict__ ln1s, const float* __restrict__ ln1b,
    const _Float16* __restrict__ W1T, const float* __restrict__ b1,
    const _Float16* __restrict__ W2T, const float* __restrict__ b2,
    const float* __restrict__ ln2s, const float* __restrict__ ln2b,
    float* __restrict__ cls_out) {
  extern __shared__ _Float16 smem[];
  _Float16* sR  = smem;                 // [64][256] residual stream (f16)
  _Float16* sQ  = smem + 16384;         // Q, later attention output O
  _Float16* sK  = smem + 2 * 16384;     // K
  _Float16* sVt = smem + 3 * 16384;     // V^T [256][64], later FFN hidden chunk [64][256]
  _Float16* sPw = smem + 4 * 16384;     // per-wave probs scratch: 8 x [16][64]
  float* sPadF  = (float*)(smem + 4 * 16384 + 8 * 1024);  // [64]

  const int b = blockIdx.x;
  const int tid = threadIdx.x;
  const int w = tid >> 5;
  const int lane = tid & 31;

  // ---- embedding sum + reference substitution ----
  {
    int s = tid & 63;
    int c0 = (tid >> 6) << 6;
    const int* tk = tokens + (b * SEQ + s) * 3;
    int t0 = tk[0], t1 = tk[1], t2 = tk[2];
    const float* e0 = emb + (long long)t0 * DMODEL;
    const float* e1 = emb + (long long)(2048 + t1) * DMODEL;
    const float* e2 = emb + (long long)(2 * 2048 + t2) * DMODEL;
    bool rm = refmask[b * SEQ + s] != 0;
    const float* rs = storage + (long long)t1 * DMODEL;
#pragma unroll 8
    for (int j = 0; j < 64; ++j) {
      int c = c0 + j;
      float v = rm ? rs[c] : (e0[c] + e1[c] + e2[c]);
      sR[s * DMODEL + c] = (_Float16)v;
    }
    if (tid < SEQ) sPadF[tid] = pad[b * SEQ + tid] ? 0.0f : NEGINF;
  }
  __syncthreads();

#pragma unroll 1
  for (int l = 0; l < NLAYER; ++l) {
    const _Float16* Wq = WqT + l * 65536;
    const _Float16* Wk = WkT + l * 65536;
    const _Float16* Wv = WvT + l * 65536;
    const _Float16* Wo = WoT + l * 65536;
    const _Float16* W1 = W1T + l * (DMODEL * FFN);
    const _Float16* W2 = W2T + l * (DMODEL * FFN);
    const float* B1 = b1 + l * FFN;
    const float* B2 = b2 + l * DMODEL;
    const float* L1s = ln1s + l * DMODEL;
    const float* L1b = ln1b + l * DMODEL;
    const float* L2s = ln2s + l * DMODEL;
    const float* L2b = ln2b + l * DMODEL;

    // ---- QKV projections: each wave owns 8 of 64 tiles ----
#pragma unroll 1
    for (int i = 0; i < 8; ++i) {
      int t = w * 8 + i;
      int mB = (t >> 4) << 4, nB = (t & 15) << 4;
      v8f q  = mma256(sR, DMODEL, Wq, DMODEL, mB, nB, lane);
      v8f kk = mma256(sR, DMODEL, Wk, DMODEL, mB, nB, lane);
      v8f vv = mma256(sR, DMODEL, Wv, DMODEL, mB, nB, lane);
      int n = nB + (lane & 15);
      int mO = mB + ((lane >> 4) << 3);
#pragma unroll
      for (int r = 0; r < 8; ++r) {
        sQ[(mO + r) * DMODEL + n] = (_Float16)q[r];
        sK[(mO + r) * DMODEL + n] = (_Float16)kk[r];
        sVt[n * SEQ + mO + r]     = (_Float16)vv[r];  // V transposed
      }
    }
    __syncthreads();

    // ---- attention: wave w = head w ----
    {
      const int h = w;
      const _Float16* Qh = sQ + h * HDIM;
      const _Float16* Kh = sK + h * HDIM;
      const _Float16* Vh = sVt + h * HDIM * SEQ;
      _Float16* Pw = sPw + w * (16 * SEQ);
      v8f oacc[8];
#pragma unroll
      for (int i = 0; i < 8; ++i) oacc[i] = vzero8();

#pragma unroll 1
      for (int mT = 0; mT < 4; ++mT) {
        v16h aq = load_a_frag(Qh, DMODEL, mT * 16, 0, lane);
        v8f sc[4];
#pragma unroll
        for (int nT = 0; nT < 4; ++nT) {
          v16h bk = load_b_frag(Kh, DMODEL, nT * 16, 0, lane);
          sc[nT] = __builtin_amdgcn_wmma_f32_16x16x32_f16(false, aq, false, bk, (short)0,
                                                          vzero8(), false, false);
        }
        const float scale = 0.17677669529663687f;  // 1/sqrt(32)
#pragma unroll
        for (int nT = 0; nT < 4; ++nT) {
          float mk = sPadF[nT * 16 + (lane & 15)];
#pragma unroll
          for (int r = 0; r < 8; ++r) sc[nT][r] = sc[nT][r] * scale + mk;
        }
        // softmax across 64 keys (4 tiles in regs + xor-shuffle within 16-lane halves)
#pragma unroll
        for (int r = 0; r < 8; ++r) {
          float mx = fmaxf(fmaxf(sc[0][r], sc[1][r]), fmaxf(sc[2][r], sc[3][r]));
#pragma unroll
          for (int off = 1; off < 16; off <<= 1) mx = fmaxf(mx, __shfl_xor(mx, off, 32));
          float e0 = __expf(sc[0][r] - mx);
          float e1 = __expf(sc[1][r] - mx);
          float e2 = __expf(sc[2][r] - mx);
          float e3 = __expf(sc[3][r] - mx);
          float sm = e0 + e1 + e2 + e3;
#pragma unroll
          for (int off = 1; off < 16; off <<= 1) sm += __shfl_xor(sm, off, 32);
          float inv = 1.0f / sm;
          int ml = ((lane >> 4) << 3) + r;
          int nc = lane & 15;
          Pw[ml * SEQ + nc]      = (_Float16)(e0 * inv);
          Pw[ml * SEQ + 16 + nc] = (_Float16)(e1 * inv);
          Pw[ml * SEQ + 32 + nc] = (_Float16)(e2 * inv);
          Pw[ml * SEQ + 48 + nc] = (_Float16)(e3 * inv);
        }
        // O row-block: P[16][64] @ V[64][32]
#pragma unroll
        for (int ks = 0; ks < 2; ++ks) {
          v16h ap = load_a_frag(Pw, SEQ, 0, ks * 32, lane);
#pragma unroll
          for (int nT = 0; nT < 2; ++nT) {
            v16h bv = load_b_frag(Vh, SEQ, nT * 16, ks * 32, lane);
            oacc[mT * 2 + nT] = __builtin_amdgcn_wmma_f32_16x16x32_f16(
                false, ap, false, bv, (short)0, oacc[mT * 2 + nT], false, false);
          }
        }
      }
      __syncthreads();  // all waves done reading sQ/sK
      // write O (f16) over sQ
#pragma unroll
      for (int mT = 0; mT < 4; ++mT) {
#pragma unroll
        for (int nT = 0; nT < 2; ++nT) {
          int n = h * HDIM + nT * 16 + (lane & 15);
          int mO = mT * 16 + ((lane >> 4) << 3);
          v8f o = oacc[mT * 2 + nT];
#pragma unroll
          for (int r = 0; r < 8; ++r) sQ[(mO + r) * DMODEL + n] = (_Float16)o[r];
        }
      }
      __syncthreads();
    }

    // ---- output projection + residual ----
#pragma unroll 1
    for (int i = 0; i < 8; ++i) {
      int t = w * 8 + i;
      int mB = (t >> 4) << 4, nB = (t & 15) << 4;
      v8f c = mma256(sQ, DMODEL, Wo, DMODEL, mB, nB, lane);
      int n = nB + (lane & 15);
      int mO = mB + ((lane >> 4) << 3);
#pragma unroll
      for (int r = 0; r < 8; ++r) {
        int idx = (mO + r) * DMODEL + n;
        sR[idx] = (_Float16)((float)sR[idx] + c[r]);
      }
    }
    __syncthreads();
    layernorm_inplace(sR, L1s, L1b, w, lane);
    __syncthreads();

    // ---- FFN: hidden in 4 chunks of 256, second GEMM accumulates in registers ----
    v8f facc[8];
#pragma unroll
    for (int i = 0; i < 8; ++i) facc[i] = vzero8();
#pragma unroll 1
    for (int ch = 0; ch < 4; ++ch) {
#pragma unroll 1
      for (int i = 0; i < 8; ++i) {
        int t = w * 8 + i;
        int mB = (t >> 4) << 4, nB = (t & 15) << 4;
        int nG = ch * 256 + nB;
        v8f c = mma256(sR, DMODEL, W1, DMODEL, mB, nG, lane);
        float bv = B1[nG + (lane & 15)];
        int mO = mB + ((lane >> 4) << 3);
#pragma unroll
        for (int r = 0; r < 8; ++r) {
          float hv = fmaxf(c[r] + bv, 0.0f);  // relu
          sVt[(mO + r) * DMODEL + nB + (lane & 15)] = (_Float16)hv;
        }
      }
      __syncthreads();
#pragma unroll
      for (int i = 0; i < 8; ++i) {
        int t = w * 8 + i;
        int mB = (t >> 4) << 4, nB = (t & 15) << 4;
        facc[i] = mma_acc_t<256>(sVt, DMODEL, W2, FFN, mB, nB, ch * 256, lane, facc[i]);
      }
      __syncthreads();
    }
#pragma unroll
    for (int i = 0; i < 8; ++i) {
      int t = w * 8 + i;
      int mB = (t >> 4) << 4, nB = (t & 15) << 4;
      int n = nB + (lane & 15);
      float bv = B2[n];
      int mO = mB + ((lane >> 4) << 3);
#pragma unroll
      for (int r = 0; r < 8; ++r) {
        int idx = (mO + r) * DMODEL + n;
        sR[idx] = (_Float16)((float)sR[idx] + facc[i][r] + bv);
      }
    }
    __syncthreads();
    layernorm_inplace(sR, L2s, L2b, w, lane);
    __syncthreads();
  }

  // CLS = row 0
  cls_out[(long long)b * DMODEL + tid] = (float)sR[tid];
}

extern "C" void kernel_launch(void* const* d_in, const int* in_sizes, int n_in,
                              void* d_out, int out_size, void* d_ws, size_t ws_size,
                              hipStream_t stream) {
  (void)in_sizes; (void)n_in; (void)out_size; (void)ws_size;
  const int NS = 2048, NH = 512;

  const int* tokens_scope = (const int*)d_in[0];
  const int* tokens_hole  = (const int*)d_in[1];
  const unsigned char* pad_scope = (const unsigned char*)d_in[2];
  const unsigned char* pad_hole  = (const unsigned char*)d_in[3];
  const unsigned char* ref_scope = (const unsigned char*)d_in[4];
  const unsigned char* ref_hole  = (const unsigned char*)d_in[5];
  const int* scope_sort = (const int*)d_in[6];
  const int* scope_pos  = (const int*)d_in[7];
  const float* emb  = (const float*)d_in[8];
  const float* Wq   = (const float*)d_in[9];
  const float* Wk   = (const float*)d_in[10];
  const float* Wv   = (const float*)d_in[11];
  const float* Wo   = (const float*)d_in[12];
  const float* ln1s = (const float*)d_in[13];
  const float* ln1b = (const float*)d_in[14];
  const float* W1   = (const float*)d_in[15];
  const float* b1   = (const float*)d_in[16];
  const float* W2   = (const float*)d_in[17];
  const float* b2   = (const float*)d_in[18];
  const float* ln2s = (const float*)d_in[19];
  const float* ln2b = (const float*)d_in[20];

  float* out_scope = (float*)d_out;
  float* out_hole  = out_scope + (long long)NS * DMODEL;

  // workspace carve-out (8 MB total)
  char* ws = (char*)d_ws;
  _Float16* WqT = (_Float16*)ws; ws += (size_t)NLAYER * 65536 * sizeof(_Float16);
  _Float16* WkT = (_Float16*)ws; ws += (size_t)NLAYER * 65536 * sizeof(_Float16);
  _Float16* WvT = (_Float16*)ws; ws += (size_t)NLAYER * 65536 * sizeof(_Float16);
  _Float16* WoT = (_Float16*)ws; ws += (size_t)NLAYER * 65536 * sizeof(_Float16);
  _Float16* W1T = (_Float16*)ws; ws += (size_t)NLAYER * DMODEL * FFN * sizeof(_Float16);
  _Float16* W2T = (_Float16*)ws; ws += (size_t)NLAYER * DMODEL * FFN * sizeof(_Float16);
  float* cls_ws = (float*)ws;

  const int tb = 256;
  int nqkv = NLAYER * DMODEL * DMODEL;
  int nffn = NLAYER * DMODEL * FFN;
  wcast_transpose<<<(nqkv + tb - 1) / tb, tb, 0, stream>>>(Wq, WqT, NLAYER, DMODEL, DMODEL);
  wcast_transpose<<<(nqkv + tb - 1) / tb, tb, 0, stream>>>(Wk, WkT, NLAYER, DMODEL, DMODEL);
  wcast_transpose<<<(nqkv + tb - 1) / tb, tb, 0, stream>>>(Wv, WvT, NLAYER, DMODEL, DMODEL);
  wcast_transpose<<<(nqkv + tb - 1) / tb, tb, 0, stream>>>(Wo, WoT, NLAYER, DMODEL, DMODEL);
  wcast_transpose<<<(nffn + tb - 1) / tb, tb, 0, stream>>>(W1, W1T, NLAYER, DMODEL, FFN);
  wcast_transpose<<<(nffn + tb - 1) / tb, tb, 0, stream>>>(W2, W2T, NLAYER, FFN, DMODEL);

  pos_enc_kernel<<<NS, 256, 0, stream>>>(scope_pos, out_scope);

  const size_t SMEM = (size_t)(4 * 16384 + 8 * 1024) * sizeof(_Float16) + 64 * sizeof(float);

  for (int rank = 0; rank < 4; ++rank) {
    encoder_kernel<<<NS, 256, SMEM, stream>>>(tokens_scope, pad_scope, ref_scope, emb, out_scope,
                                              WqT, WkT, WvT, WoT, ln1s, ln1b, W1T, b1, W2T, b2,
                                              ln2s, ln2b, cls_ws);
    rank_update_kernel<<<NS, 256, 0, stream>>>(scope_sort, cls_ws, out_scope, rank);
  }
  encoder_kernel<<<NH, 256, SMEM, stream>>>(tokens_hole, pad_hole, ref_hole, emb, out_scope,
                                            WqT, WkT, WvT, WoT, ln1s, ln1b, W1T, b1, W2T, b2,
                                            ln2s, ln2b, out_hole);
}
